// PolynomialLayer_57810259804463
// MI455X (gfx1250) — matrix operand order repaired and checked by reading, not compiled, not run on hardware
//
#include <hip/hip_runtime.h>

typedef __bf16 v16bf __attribute__((ext_vector_type(16)));
typedef float  v8f   __attribute__((ext_vector_type(8)));
typedef unsigned int u32x4 __attribute__((ext_vector_type(4)));
typedef int i32x4 __attribute__((ext_vector_type(4)));
typedef int i32x8 __attribute__((ext_vector_type(8)));

#if defined(__AMDGCN__) && __has_builtin(__builtin_amdgcn_tensor_load_to_lds)
#define HAVE_TDM_BUILTIN 1
#warning "gfx1250 device pass: tensor_load_to_lds BUILTIN enabled"
#else
#define HAVE_TDM_BUILTIN 0
#endif

#if defined(__AMDGCN__) && __has_builtin(__builtin_amdgcn_cvt_pk_bf16_f32)
#define HAVE_PK 1
#warning "gfx1250 device pass: cvt_pk_bf16_f32 BUILTIN enabled"
#else
#define HAVE_PK 0
#endif

#define D_IN   128
#define K_TOT  8384
#define N_TOT  512
#define B_TOT  8192
#define MT     128          // M tile per workgroup
#define NT     128          // N tile per workgroup (8 subtiles per wave)
#define KC     32           // K chunk (one bf16 WMMA K)
#define NPOLY  8256
#define XPADW  132          // padded x-tile row stride (f32 units)

__device__ __forceinline__ unsigned short f2bf(float f) {
    unsigned int u = __builtin_bit_cast(unsigned int, f);
    u += 0x7FFFu + ((u >> 16) & 1u);
    return (unsigned short)(u >> 16);
}

__device__ __forceinline__ unsigned int pack2bf(float a, float b) {
#if HAVE_PK
    typedef __bf16 v2bf __attribute__((ext_vector_type(2)));
    v2bf r = __builtin_amdgcn_cvt_pk_bf16_f32(a, b);
    return __builtin_bit_cast(unsigned int, r);
#elif defined(__AMDGCN__)
    unsigned int d;
    asm("v_cvt_pk_bf16_f32 %0, %1, %2" : "=v"(d) : "v"(a), "v"(b));
    return d;
#else
    return (unsigned int)f2bf(a) | ((unsigned int)f2bf(b) << 16);
#endif
}

// ---- o -> (i,j) pair table for degree-2 terms; S(i) = i*(257-i)/2 ----
__global__ void poly_pair_init(unsigned short* __restrict__ tbl) {
    int t = blockIdx.x * blockDim.x + threadIdx.x;
    if (t >= NPOLY) return;
    int i = (int)((257.0f - sqrtf(66049.0f - 8.0f * (float)t)) * 0.5f);
    if (i < 0) i = 0;
    if (i > 127) i = 127;
    while (i > 0 && (i * (257 - i)) / 2 > t) --i;
    while (i < 127 && ((i + 1) * (257 - (i + 1))) / 2 <= t) ++i;
    int j = i + (t - (i * (257 - i)) / 2);
    tbl[t] = (unsigned short)(i | (j << 8));
}

// ---- weights f32 -> bf16 in per-lane WMMA B-fragment layout:
// dword idx = (((kc*32 + ntile)*32 + lane)*8 + r) holds w[n][k],w[n][k+1] where
// n = ntile*16 + (lane&15), k = kc*32 + (lane>>4)*16 + 2r.
__global__ void pack_weights(const float* __restrict__ w,
                             unsigned int* __restrict__ wb) {
    unsigned int idx = blockIdx.x * 256 + threadIdx.x;
    if (idx >= (unsigned)(N_TOT * (K_TOT / 2))) return;
    int r    = idx & 7;
    int lane = (idx >> 3) & 31;
    int nt   = (idx >> 8) & 31;          // N_TOT/16 = 32 tiles
    int kc   = idx >> 13;                // 8192 dwords per k-chunk
    int n = nt * 16 + (lane & 15);
    int k = kc * KC + (lane >> 4) * 16 + 2 * r;
    const float* p = w + (size_t)n * K_TOT + k;
    wb[idx] = pack2bf(p[0], p[1]);
}

__global__ __launch_bounds__(256) void poly_wmma_gemm(
    const float* __restrict__ x, const unsigned int* __restrict__ wb,
    const float* __restrict__ bias, const unsigned short* __restrict__ pairs,
    float* __restrict__ out)
{
    __shared__ float        xs[MT * XPADW];        // x tile f32, padded (67.6 KB)
    __shared__ unsigned int fs32[MT * (KC / 2)];   // feats chunk bf16 (8 KB)

    const int tid  = threadIdx.x;
    const int lane = tid & 31;
    const int wv   = tid >> 5;
    const int m0   = blockIdx.x * MT;
    const int n0   = blockIdx.y * NT;

    // ---- stage x tile via Tensor Data Mover (HW row padding -> stride 132) ----
#if defined(__AMDGCN__)
    if (wv == 0) {
        unsigned long long ga = (unsigned long long)(const void*)(x + (size_t)m0 * D_IN);
        unsigned int lds_a = (unsigned int)(unsigned long long)(void*)xs;
        u32x4 g0;
        g0[0] = 1u;                                        // count=1
        g0[1] = lds_a;
        g0[2] = (unsigned int)(ga & 0xFFFFFFFFu);
        g0[3] = (unsigned int)((ga >> 32) & 0x01FFFFFFu) | (2u << 30); // type=2
        i32x8 g1;
        g1[0] = (int)((2u << 16)       // data_size = 4B
               | (1u << 20)            // pad_enable
               | (6u << 22)            // pad_interval: 128 dwords
               | (3u << 25));          // pad_amount: 4 dwords -> row stride 132
        g1[1] = (int)((unsigned)D_IN << 16);     // tensor_dim0 lo16 = 128
        g1[2] = (int)((unsigned)MT << 16);       // dim0 hi=0 | tensor_dim1 lo16
        g1[3] = (int)((unsigned)D_IN << 16);     // dim1 hi=0 | tile_dim0 = 128
        g1[4] = MT;                              // tile_dim1 = 128 | tile_dim2 = 0
        g1[5] = D_IN;                            // tensor_dim0_stride = 128
        g1[6] = 0;
        g1[7] = 0;
        i32x4 g2 = {0, 0, 0, 0};
        i32x4 g3 = {0, 0, 0, 0};
#if HAVE_TDM_BUILTIN
#if __clang_major__ >= 23
        i32x8 gpad = {0, 0, 0, 0, 0, 0, 0, 0};
        __builtin_amdgcn_tensor_load_to_lds(g0, g1, g2, g3, gpad, 0);
#else
        __builtin_amdgcn_tensor_load_to_lds(g0, g1, g2, g3, 0);
#endif
#else
        asm volatile("tensor_load_to_lds %0, %1, %2, %3"
                     :: "s"(g0), "s"(g1), "s"(g2), "s"(g3) : "memory");
#endif
        __builtin_amdgcn_s_wait_tensorcnt(0);
    }
#else
    // host-pass placeholder: plain staging (never codegen'd for device)
    for (int it = tid; it < MT * (D_IN / 4); it += 256) {
        int m = it >> 5, c4 = it & 31;
        *(float4*)&xs[m * XPADW + c4 * 4] = ((const float4*)x)[(size_t)(m0 + m) * 32 + c4];
    }
#endif

    v8f acc[8];
    {
        v8f z = {0.f, 0.f, 0.f, 0.f, 0.f, 0.f, 0.f, 0.f};
        #pragma unroll
        for (int i = 0; i < 8; ++i) acc[i] = z;
    }
    __syncthreads();

    const int    fm    = tid >> 1;                // feats row this thread fills
    const int    cbase = (tid & 1) * 16;          // feats col base
    const float* xrow  = &xs[fm * XPADW];
    unsigned int* frow = &fs32[fm * (KC / 2) + (cbase >> 1)];
    const unsigned int* arow =
        (const unsigned int*)&fs32[(wv * 16 + (lane & 15)) * (KC / 2)];
    const int hb = (lane >> 4) * 8;
    const unsigned int* p32 = (const unsigned int*)pairs;

    for (int k0 = 0; k0 < K_TOT; k0 += KC) {
        // ---- feats chunk 128 x 32 bf16 (fused polynomial expansion) ----
        if (k0 < D_IN) {                          // pure linear-term chunks
            #pragma unroll
            for (int cp = 0; cp < 8; ++cp) {
                int o0 = k0 + cbase + 2 * cp;
                frow[cp] = pack2bf(xrow[o0], xrow[o0 + 1]);
            }
        } else {                                  // pure degree-2 chunks
            #pragma unroll
            for (int cp = 0; cp < 8; ++cp) {
                int o0 = k0 + cbase + 2 * cp;
                unsigned int pp = p32[(o0 - D_IN) >> 1];   // 2 packed (i,j) pairs
                float a0 = xrow[pp & 0xFF]         * xrow[(pp >> 8) & 0xFF];
                float a1 = xrow[(pp >> 16) & 0xFF] * xrow[pp >> 24];
                frow[cp] = pack2bf(a0, a1);
            }
        }
        __syncthreads();

        // ---- A fragment from LDS (lane = M, K pairs per VGPR) ----
        union { unsigned int u[8]; v16bf v; } afr;
        #pragma unroll
        for (int r = 0; r < 4; ++r) afr.u[r]     = arow[hb + r];       // K 16h+0..7
        #pragma unroll
        for (int r = 0; r < 4; ++r) afr.u[4 + r] = arow[4 + hb + r];   // K 16h+8..15

        // ---- 8 N-subtiles: pre-packed B fragments straight from L2 ----
        const unsigned int* wbase =
            wb + (((size_t)(k0 >> 5) * 32 + (n0 >> 4)) * 32 + lane) * 8;
        #pragma unroll
        for (int ns = 0; ns < 8; ++ns) {
            union { u32x4 q[2]; unsigned int u[8]; v16bf v; } bfr;
            const u32x4* bp = (const u32x4*)(wbase + ns * 256);
            bfr.q[0] = bp[0];
            bfr.q[1] = bp[1];
            acc[ns] = __builtin_amdgcn_wmma_f32_16x16x32_bf16(
                false, afr.v, false, bfr.v, (short)0, acc[ns], false, false);
        }
        __syncthreads();
    }

    // ---- epilogue: bias add + coalesced store (C layout: lane=N, VGPR r=M) ----
    {
        const int nl = lane & 15;
        const int mb = (lane >> 4) * 8;
        #pragma unroll
        for (int ns = 0; ns < 8; ++ns) {
            int n = n0 + ns * 16 + nl;
            float bv = bias[n];
            #pragma unroll
            for (int r = 0; r < 8; ++r) {
                int m = m0 + wv * 16 + mb + r;
                out[(size_t)m * N_TOT + n] = acc[ns][r] + bv;
            }
        }
    }
}

// ---- exact-f32 fallback (only if ws_size cannot hold packed weights) ----
__global__ void poly_naive(const float* __restrict__ x, const float* __restrict__ w,
                           const float* __restrict__ bias, float* __restrict__ out) {
    unsigned int idx = blockIdx.x * 256 + threadIdx.x;
    if (idx >= (unsigned)B_TOT * N_TOT) return;
    int b = idx >> 9;                 // / N_TOT
    int n = idx & (N_TOT - 1);
    const float* xr = x + (size_t)b * D_IN;
    const float* wr = w + (size_t)n * K_TOT;
    float a = bias[n];
    for (int d = 0; d < D_IN; ++d) a += xr[d] * wr[d];
    int o = D_IN;
    for (int i = 0; i < D_IN; ++i) {
        float xi = xr[i];
        for (int j = i; j < D_IN; ++j, ++o) a += xi * xr[j] * wr[o];
    }
    out[idx] = a;
}

extern "C" void kernel_launch(void* const* d_in, const int* in_sizes, int n_in,
                              void* d_out, int out_size, void* d_ws, size_t ws_size,
                              hipStream_t stream) {
    const float* x    = (const float*)d_in[0];
    const float* wgt  = (const float*)d_in[1];
    const float* bias = (const float*)d_in[2];

    const size_t WB_DWORDS = (size_t)N_TOT * (K_TOT / 2);   // 2,146,304
    const size_t WB_BYTES  = WB_DWORDS * 4;                 // 8,585,216
    const size_t NEED      = WB_BYTES + (size_t)NPOLY * 2;

    if (ws_size >= NEED) {
        unsigned int*   wbuf = (unsigned int*)d_ws;
        unsigned short* tbl  = (unsigned short*)((char*)d_ws + WB_BYTES);
        poly_pair_init<<<(NPOLY + 255) / 256, 256, 0, stream>>>(tbl);
        pack_weights<<<(int)((WB_DWORDS + 255) / 256), 256, 0, stream>>>(wgt, wbuf);
        dim3 grid(B_TOT / MT, N_TOT / NT);                  // 64 x 4
        poly_wmma_gemm<<<grid, 256, 0, stream>>>(x, wbuf, bias, tbl, (float*)d_out);
    } else {
        int total = B_TOT * N_TOT;
        poly_naive<<<(total + 255) / 256, 256, 0, stream>>>(x, wgt, bias,
                                                            (float*)d_out);
    }
    (void)in_sizes; (void)n_in; (void)out_size;
}
